// RelativeMultiHeadAttention_3839700763057
// MI455X (gfx1250) — matrix operand order repaired and checked by reading, not compiled, not run on hardware
//
#include <hip/hip_runtime.h>
#include <stdint.h>
#include <stddef.h>

// ---------------------------------------------------------------------------
// RelativeMultiHeadAttention for MI455X (gfx1250), wave32 + WMMA bf16.
// ---------------------------------------------------------------------------

typedef __attribute__((ext_vector_type(16))) __bf16 v16bf;
typedef __attribute__((ext_vector_type(8)))  float  v8f;
typedef __attribute__((ext_vector_type(4)))  int    i4vec;

#define T_LEN   1024
#define D_MODEL 512
#define NHEAD   8
#define DHEAD   64
#define KDIM    512

#define AS1 __attribute__((address_space(1)))
#define AS3 __attribute__((address_space(3)))

#if __has_builtin(__builtin_amdgcn_global_load_async_to_lds_b128)
#define HAVE_ASYNC_LDS 1
#else
#define HAVE_ASYNC_LDS 0
#endif

#if __has_builtin(__builtin_amdgcn_cvt_pk_bf16_f32)
#define HAVE_PK_BF16 1
typedef __attribute__((ext_vector_type(2))) __bf16 v2bf;
#else
#define HAVE_PK_BF16 0
#endif

union V16U { v16bf v; unsigned short u[16]; };

__device__ __forceinline__ unsigned short f2bf(float f) {
  union { float f; unsigned u; } x; x.f = f;
  unsigned r = x.u + 0x7FFFu + ((x.u >> 16) & 1u);   // round-to-nearest-even
  return (unsigned short)(r >> 16);
}
__device__ __forceinline__ float bf2f(unsigned short s) {
  union { unsigned u; float f; } x; x.u = ((unsigned)s) << 16; return x.f;
}
// packed f32x2 -> bf16x2 (hardware cvt when available)
__device__ __forceinline__ void pack2(unsigned short* dst, float a, float b) {
#if HAVE_PK_BF16
  union { v2bf v; unsigned short u[2]; } t;
  t.v = __builtin_amdgcn_cvt_pk_bf16_f32(a, b);
  dst[0] = t.u[0]; dst[1] = t.u[1];
#else
  dst[0] = f2bf(a); dst[1] = f2bf(b);
#endif
}

// A fragment (16x32, 16-bit): lane = M row; lane<16 holds K kb..kb+15,
// lane>=16 holds K kb+16..kb+31  (contiguous order, matches B fragments)
__device__ __forceinline__ v16bf a_frag_f32(const float* __restrict__ rowp, int kb, int hl) {
  V16U r; int o0 = kb + hl * 16, o1 = kb + 8 + hl * 16;
#pragma unroll
  for (int j = 0; j < 8; j += 2) {
    pack2(&r.u[j],     rowp[o0 + j], rowp[o0 + j + 1]);
    pack2(&r.u[8 + j], rowp[o1 + j], rowp[o1 + j + 1]);
  }
  return r.v;
}
__device__ __forceinline__ v16bf a_frag_bf(const unsigned short* rowp, int kb, int hl) {
  V16U r; int o0 = kb + hl * 16, o1 = kb + 8 + hl * 16;
#pragma unroll
  for (int j = 0; j < 8; ++j) { r.u[j] = rowp[o0 + j]; r.u[8 + j] = rowp[o1 + j]; }
  return r.v;
}
// B fragment (32x16, 16-bit): lane = N col; lane<16 elements = K kb..kb+15,
// lane>=16 elements = K kb+16..kb+31. Caller passes p = col_ptr + kb + (lane>=16)*16.
__device__ __forceinline__ v16bf b_frag_bf(const unsigned short* p) {
  V16U r;
#pragma unroll
  for (int j = 0; j < 16; ++j) r.u[j] = p[j];
  return r.v;
}

#define WMMA_BF16(a, b, c) \
  __builtin_amdgcn_wmma_f32_16x16x32_bf16(false, (a), false, (b), (short)0, (c), false, false)

// ---------------------------------------------------------------------------
// Weight transpose + f32->bf16: Wt[n][k] = W[k][n]   (512x512)
// ---------------------------------------------------------------------------
__global__ void __launch_bounds__(1024) wt_cvt(const float* __restrict__ W,
                                               unsigned short* __restrict__ Wt) {
  __shared__ float tile[32][33];
  const int tx = threadIdx.x, ty = threadIdx.y;
  const int bx = blockIdx.x * 32, by = blockIdx.y * 32;
  tile[ty][tx] = W[(size_t)(by + ty) * KDIM + bx + tx];
  __syncthreads();
  Wt[(size_t)(bx + ty) * KDIM + by + tx] = f2bf(tile[tx][ty]);
}

// ---------------------------------------------------------------------------
// Projection GEMM: out = A(f32)[M x 512] @ Wt^T + bias (+per-column biases).
// layout 0: out[((b*H+h)*T + t)*DK + d]   (Q/K/P)
// layout 1: out[((b*H+h)*DK + d)*T + t]   (V transposed for attn*V B-frags)
// ---------------------------------------------------------------------------
__global__ void __launch_bounds__(512) proj_gemm(
    const float* __restrict__ A, const unsigned short* __restrict__ Wt,
    const float* __restrict__ bias, const float* __restrict__ cb0,
    const float* __restrict__ cb1,
    unsigned short* __restrict__ out0, unsigned short* __restrict__ out1,
    int layout) {
  const int tid = threadIdx.x;
  const int wave = tid >> 5, lane = tid & 31, l16 = lane & 15, hl = lane >> 4;
  const int wm = wave >> 2, wn = wave & 3;
  const int m0 = blockIdx.y * 64, n0 = blockIdx.x * 64;
  const int m = m0 + 16 * wm + l16;
  const int n = n0 + 16 * wn + l16;
  const float* arow = A + (size_t)m * KDIM;
  const unsigned short* wcol = Wt + (size_t)n * KDIM;

  v8f acc = {};
#pragma unroll 4
  for (int kk = 0; kk < KDIM; kk += 32) {
    v16bf a = a_frag_f32(arow, kk, hl);
    v16bf b = b_frag_bf(wcol + kk + hl * 16);
    acc = WMMA_BF16(a, b, acc);
  }
  const float bsum = bias ? bias[n] : 0.f;
  const float c0 = cb0 ? cb0[n] : 0.f;
  const float c1 = cb1 ? cb1[n] : 0.f;
  const int h = n / DHEAD, d = n % DHEAD;
#pragma unroll
  for (int r = 0; r < 8; ++r) {
    const int mm = m0 + 16 * wm + r + hl * 8;
    const int b_ = mm / T_LEN, t = mm % T_LEN;
    const float v = acc[r] + bsum;
    size_t idx;
    if (layout == 0) idx = ((size_t)(b_ * NHEAD + h) * T_LEN + t) * DHEAD + d;
    else             idx = ((size_t)(b_ * NHEAD + h) * DHEAD + d) * T_LEN + t;
    out0[idx] = f2bf(v + c0);
    if (out1) out1[idx] = f2bf(v + c1);
  }
}

// ---------------------------------------------------------------------------
// BD = Qv * P^T per (b,h), scattered through the legacy rel_shift permutation
// into pre-zeroed Sws[b,h,q,k] (bf16). Map: g=t(T+1)+p+1, q=g/T-1, k=g%T.
// ---------------------------------------------------------------------------
__global__ void __launch_bounds__(512) bd_gemm(
    const unsigned short* __restrict__ Qv, const unsigned short* __restrict__ Pp,
    unsigned short* __restrict__ Sws) {
  const int tid = threadIdx.x;
  const int wave = tid >> 5, lane = tid & 31, l16 = lane & 15, hl = lane >> 4;
  const int wt_ = wave >> 2, wp = wave & 3;
  const int p0 = blockIdx.x * 64, t0 = blockIdx.y * 64;
  const int bh = blockIdx.z;
  const int h = bh % NHEAD;
  const unsigned short* qbase = Qv + (size_t)bh * T_LEN * DHEAD;
  const unsigned short* pbase = Pp + (size_t)h * T_LEN * DHEAD;
  unsigned short* sbase = Sws + (size_t)bh * T_LEN * T_LEN;

  const int t = t0 + 16 * wt_ + l16;
  const int p = p0 + 16 * wp + l16;
  const unsigned short* arow = qbase + (size_t)t * DHEAD;
  const unsigned short* bcol = pbase + (size_t)p * DHEAD;

  v8f acc = {};
#pragma unroll
  for (int c = 0; c < 2; ++c) {
    v16bf a = a_frag_bf(arow, 32 * c, hl);
    v16bf b = b_frag_bf(bcol + 32 * c + hl * 16);
    acc = WMMA_BF16(a, b, acc);
  }
#pragma unroll
  for (int r = 0; r < 8; ++r) {
    const int tt = t0 + 16 * wt_ + r + hl * 8;
    const int pc = p0 + 16 * wp + l16;
    const int g = tt * (T_LEN + 1) + pc + 1;
    const int q = g / T_LEN - 1;
    const int k = g - (q + 1) * T_LEN;
    if (q >= 0) sbase[(size_t)q * T_LEN + k] = f2bf(acc[r]);
  }
}

// ---------------------------------------------------------------------------
// Flash attention with online softmax. One block per (b,h,64 q-rows),
// 16 waves; wave (wq,wk) owns a 16x16 subtile of both S and O.
// K/V tiles staged to LDS with GLOBAL_LOAD_ASYNC_TO_LDS_B128, double-buffered.
// ---------------------------------------------------------------------------
__global__ void __launch_bounds__(512) attn_kernel(
    const unsigned short* __restrict__ Qu, const unsigned short* __restrict__ Kb,
    const unsigned short* __restrict__ Vt, const unsigned short* __restrict__ Sws,
    unsigned short* __restrict__ Xb) {
  __shared__ unsigned short p_lds[64 * 64];
#if HAVE_ASYNC_LDS
  __shared__ unsigned short k_lds[2][64 * 64];
  __shared__ unsigned short v_lds[2][64 * 64];
#endif
  __shared__ float red_max[4][64];
  __shared__ float red_sum[4][64];
  __shared__ float m_run[64], l_run[64], m_new_s[64], sc_s[64];

  const int tid = threadIdx.x;
  const int wave = tid >> 5, lane = tid & 31, l16 = lane & 15, hl = lane >> 4;
  const int wq = wave >> 2, wk = wave & 3;
  const int q0 = blockIdx.x * 64;
  const int bh = blockIdx.y;
  const int b = bh / NHEAD, h = bh % NHEAD;
  const unsigned short* qbase = Qu + (size_t)bh * T_LEN * DHEAD;
  const unsigned short* kbase = Kb + (size_t)bh * T_LEN * DHEAD;
  const unsigned short* vbase = Vt + (size_t)bh * DHEAD * T_LEN;
  const unsigned short* sbase = Sws + (size_t)bh * T_LEN * T_LEN;

  if (tid < 64) { m_run[tid] = -1e30f; l_run[tid] = 0.f; }

#if HAVE_ASYNC_LDS
  // Each of 512 threads copies one 16B chunk of the 64x64 bf16 K and V tiles.
  const int srow = tid >> 3;            // 0..63
  const int scol = (tid & 7) * 8;       // ushort offset within the row
  auto stage = [&](int kt2, int bsel) {
    const unsigned short* gk = kbase + (size_t)(kt2 * 64 + srow) * DHEAD + scol;
    const unsigned short* gv = vbase + (size_t)srow * T_LEN + kt2 * 64 + scol;
    __builtin_amdgcn_global_load_async_to_lds_b128(
        (AS1 i4vec*)gk, (AS3 i4vec*)&k_lds[bsel][srow * 64 + scol], 0, 0);
    __builtin_amdgcn_global_load_async_to_lds_b128(
        (AS1 i4vec*)gv, (AS3 i4vec*)&v_lds[bsel][srow * 64 + scol], 0, 0);
  };
  stage(0, 0);
  asm volatile("s_wait_asynccnt 0x0" ::: "memory");
#endif
  __syncthreads();

  // Qu A-fragments for this wave's 16 query rows (DK=64 -> two K=32 chunks)
  const unsigned short* qrow = qbase + (size_t)(q0 + 16 * wq + l16) * DHEAD;
  const v16bf aq0 = a_frag_bf(qrow, 0, hl);
  const v16bf aq1 = a_frag_bf(qrow, 32, hl);

  v8f o_acc = {};
  const float scaling = 0.125f;  // DK^-0.5
  int buf = 0;

  for (int kt = 0; kt < T_LEN / 64; ++kt) {
    const int k0 = kt * 64;
#if HAVE_ASYNC_LDS
    if (kt + 1 < T_LEN / 64) stage(kt + 1, buf ^ 1);   // overlap copy with compute
    const unsigned short* krow = &k_lds[buf][(16 * wk + l16) * 64];
    const unsigned short* vrow = &v_lds[buf][(16 * wk + l16) * 64];
#else
    const unsigned short* krow = kbase + (size_t)(k0 + 16 * wk + l16) * DHEAD;
    const unsigned short* vrow = vbase + (size_t)(16 * wk + l16) * T_LEN + k0;
    __builtin_prefetch(krow + 64 * DHEAD, 0, 0);       // -> global_prefetch_b8
    __builtin_prefetch(vrow + 64, 0, 0);
#endif
    // ---- S = Qu * K^T ----
    v8f s = {};
    s = WMMA_BF16(aq0, b_frag_bf(krow + hl * 16), s);
    s = WMMA_BF16(aq1, b_frag_bf(krow + 32 + hl * 16), s);

    // add shifted positional scores, scale
    const int kcol = k0 + 16 * wk + l16;
#pragma unroll
    for (int r = 0; r < 8; ++r) {
      const int qq = q0 + 16 * wq + r + hl * 8;
      const float bd = bf2f(sbase[(size_t)qq * T_LEN + kcol]);
      s[r] = (s[r] + bd) * scaling;
    }
    // per-row max within this wave's 16 columns (rows r / r+8 in 16-lane halves)
#pragma unroll
    for (int r = 0; r < 8; ++r) {
      float v = s[r];
      v = fmaxf(v, __shfl_xor(v, 1, 16));
      v = fmaxf(v, __shfl_xor(v, 2, 16));
      v = fmaxf(v, __shfl_xor(v, 4, 16));
      v = fmaxf(v, __shfl_xor(v, 8, 16));
      if (l16 == 0) red_max[wk][16 * wq + r + hl * 8] = v;
    }
    __syncthreads();
    if (tid < 64) {
      const float mt = fmaxf(fmaxf(red_max[0][tid], red_max[1][tid]),
                             fmaxf(red_max[2][tid], red_max[3][tid]));
      const float mo = m_run[tid];
      const float mn = fmaxf(mo, mt);
      m_new_s[tid] = mn;
      sc_s[tid] = __expf(mo - mn);
      m_run[tid] = mn;
    }
    __syncthreads();
    // P = exp(S - m_new), rescale O, row sums, stash P as bf16 in LDS
#pragma unroll
    for (int r = 0; r < 8; ++r) {
      const int row = 16 * wq + r + hl * 8;
      const float p = __expf(s[r] - m_new_s[row]);
      o_acc[r] *= sc_s[row];
      float v = p;
      v += __shfl_xor(v, 1, 16);
      v += __shfl_xor(v, 2, 16);
      v += __shfl_xor(v, 4, 16);
      v += __shfl_xor(v, 8, 16);
      if (l16 == 0) red_sum[wk][row] = v;
      p_lds[row * 64 + 16 * wk + l16] = f2bf(p);
    }
    __syncthreads();
    if (tid < 64) {
      l_run[tid] = l_run[tid] * sc_s[tid] +
                   red_sum[0][tid] + red_sum[1][tid] + red_sum[2][tid] + red_sum[3][tid];
    }
    // ---- O += P * V  (wave role becomes (wq, wd=wk)) ----
    const unsigned short* prow = &p_lds[(16 * wq + l16) * 64];
    o_acc = WMMA_BF16(a_frag_bf(prow, 0, hl),  b_frag_bf(vrow + hl * 16),      o_acc);
    o_acc = WMMA_BF16(a_frag_bf(prow, 32, hl), b_frag_bf(vrow + 32 + hl * 16), o_acc);
#if HAVE_ASYNC_LDS
    asm volatile("s_wait_asynccnt 0x0" ::: "memory");  // next tile landed
#endif
    __syncthreads();   // publish staged tiles; also protects p_lds reuse
    buf ^= 1;
  }
  // normalize + write X[b, t, h*DK + d] (bf16, GEMM-ready row-major)
  const int d = 16 * wk + l16;
#pragma unroll
  for (int r = 0; r < 8; ++r) {
    const int row = 16 * wq + r + hl * 8;
    const float v = o_acc[r] / l_run[row];
    Xb[((size_t)b * T_LEN + (q0 + row)) * D_MODEL + h * DHEAD + d] = f2bf(v);
  }
}

// ---------------------------------------------------------------------------
// Output GEMM: out(f32) = X(bf16)[M x 512] @ Wo + bo
// ---------------------------------------------------------------------------
__global__ void __launch_bounds__(512) out_gemm(
    const unsigned short* __restrict__ A, const unsigned short* __restrict__ Wt,
    const float* __restrict__ bias, float* __restrict__ out) {
  const int tid = threadIdx.x;
  const int wave = tid >> 5, lane = tid & 31, l16 = lane & 15, hl = lane >> 4;
  const int wm = wave >> 2, wn = wave & 3;
  const int m0 = blockIdx.y * 64, n0 = blockIdx.x * 64;
  const int m = m0 + 16 * wm + l16;
  const int n = n0 + 16 * wn + l16;
  const unsigned short* arow = A + (size_t)m * KDIM;
  const unsigned short* wcol = Wt + (size_t)n * KDIM;

  v8f acc = {};
#pragma unroll 4
  for (int kk = 0; kk < KDIM; kk += 32) {
    v16bf a = a_frag_bf(arow, kk, hl);
    v16bf b = b_frag_bf(wcol + kk + hl * 16);
    acc = WMMA_BF16(a, b, acc);
  }
  const float bsum = bias[n];
#pragma unroll
  for (int r = 0; r < 8; ++r) {
    const int mm = m0 + 16 * wm + r + hl * 8;
    out[(size_t)mm * D_MODEL + n] = acc[r] + bsum;
  }
}

// ---------------------------------------------------------------------------
extern "C" void kernel_launch(void* const* d_in, const int* in_sizes, int n_in,
                              void* d_out, int out_size, void* d_ws, size_t ws_size,
                              hipStream_t stream) {
  (void)n_in; (void)out_size; (void)ws_size;
  const float* query = (const float*)d_in[0];
  const float* key   = (const float*)d_in[1];
  const float* value = (const float*)d_in[2];
  const float* pos   = (const float*)d_in[3];
  const float* Wq = (const float*)d_in[4];   const float* bq = (const float*)d_in[5];
  const float* Wk = (const float*)d_in[6];   const float* bk = (const float*)d_in[7];
  const float* Wv = (const float*)d_in[8];   const float* bv = (const float*)d_in[9];
  const float* Wo = (const float*)d_in[10];  const float* bo = (const float*)d_in[11];
  const float* Wpos = (const float*)d_in[12];
  const float* pbu  = (const float*)d_in[13];
  const float* pbv  = (const float*)d_in[14];

  const int T = T_LEN, D = D_MODEL, H = NHEAD;
  const int B = in_sizes[0] / (T * D);   // 4

  char* ws = (char*)d_ws;
  size_t off = 0;
  auto take = [&](size_t bytes) -> char* {
    char* p = ws + off; off = (off + bytes + 255) & ~(size_t)255; return p;
  };
  unsigned short* sws = (unsigned short*)take((size_t)B * H * T * T * 2);  // 64 MB
  unsigned short* qu  = (unsigned short*)take((size_t)B * T * D * 2);
  unsigned short* qv  = (unsigned short*)take((size_t)B * T * D * 2);
  unsigned short* kb  = (unsigned short*)take((size_t)B * T * D * 2);
  unsigned short* vt  = (unsigned short*)take((size_t)B * T * D * 2);
  unsigned short* pp  = (unsigned short*)take((size_t)T * D * 2);
  unsigned short* xb  = (unsigned short*)take((size_t)B * T * D * 2);
  unsigned short* wtq = (unsigned short*)take((size_t)D * D * 2);
  unsigned short* wtk = (unsigned short*)take((size_t)D * D * 2);
  unsigned short* wtv = (unsigned short*)take((size_t)D * D * 2);
  unsigned short* wtp = (unsigned short*)take((size_t)D * D * 2);
  unsigned short* wto = (unsigned short*)take((size_t)D * D * 2);

  (void)hipMemsetAsync(sws, 0, (size_t)B * H * T * T * 2, stream);  // zero rel-shift target

  dim3 tb(32, 32);
  wt_cvt<<<dim3(16, 16), tb, 0, stream>>>(Wq,   wtq);
  wt_cvt<<<dim3(16, 16), tb, 0, stream>>>(Wk,   wtk);
  wt_cvt<<<dim3(16, 16), tb, 0, stream>>>(Wv,   wtv);
  wt_cvt<<<dim3(16, 16), tb, 0, stream>>>(Wpos, wtp);
  wt_cvt<<<dim3(16, 16), tb, 0, stream>>>(Wo,   wto);

  dim3 blk(512, 1, 1);
  // Q: write q+bias+u and q+bias+v in one pass (column biases)
  proj_gemm<<<dim3(D / 64, B * T / 64), blk, 0, stream>>>(query, wtq, bq, pbu, pbv, qu, qv, 0);
  proj_gemm<<<dim3(D / 64, B * T / 64), blk, 0, stream>>>(key,   wtk, bk, nullptr, nullptr, kb, nullptr, 0);
  proj_gemm<<<dim3(D / 64, B * T / 64), blk, 0, stream>>>(value, wtv, bv, nullptr, nullptr, vt, nullptr, 1);
  proj_gemm<<<dim3(D / 64, T / 64),     blk, 0, stream>>>(pos,   wtp, nullptr, nullptr, nullptr, pp, nullptr, 0);

  bd_gemm<<<dim3(T / 64, T / 64, B * H), blk, 0, stream>>>(qv, pp, sws);
  attn_kernel<<<dim3(T / 64, B * H),     blk, 0, stream>>>(qu, kb, vt, sws, xb);
  out_gemm<<<dim3(D / 64, B * T / 64),   blk, 0, stream>>>(xb, wto, bo, (float*)d_out);
}
